// TransformerBlock_70188355551258
// MI455X (gfx1250) — matrix operand order repaired and checked by reading, not compiled, not run on hardware
//
#include <hip/hip_runtime.h>
#include <hip/hip_bf16.h>

// ---------------------------------------------------------------------------
// Types for CDNA5 WMMA (wave32, 16x16x32 f16 -> f32)
// ---------------------------------------------------------------------------
typedef __attribute__((ext_vector_type(16))) _Float16 v16h;
typedef __attribute__((ext_vector_type(8)))  float    v8f;

union F16x16 { v16h h; unsigned u[8]; };
union F32x8  { v8f  v; float    f[8]; };

__device__ __forceinline__ v8f wmma_f16(v16h a, v16h b, v8f c) {
  // D = A(16x32,f16) * B(32x16,f16) + C(16x16,f32)
  return __builtin_amdgcn_wmma_f32_16x16x32_f16(false, a, false, b,
                                                (short)0, c, false, false);
}

// ---------------------------------------------------------------------------
// Problem constants
// ---------------------------------------------------------------------------
#define BB    8
#define TT    1024
#define HH    16
#define DD    64
#define LL    256
#define EE    8
#define II    2048
#define CC    1024
#define NTOK  8192          // B*T
#define NPAD  9216          // 72 tiles * 128 rows (padded expert segments)
#define NTILE 72

// ---------------------------------------------------------------------------
// Generic WMMA GEMM: C[M,N] = alpha * A[M,K] * B[K,N] (+Res), f32 in/out,
// f16 compute, f32 accumulate. Block tile 128 x BN (BN = 128 or 64),
// 8 waves in a 4x2 grid, each wave a 32 x BN/2 sub-tile (2 x FN fragments).
// Register double-buffering: next A/B tiles prefetched into VGPRs while the
// WMMAs consume the current LDS tile. Compile-time specialization removes
// gather/scatter/expert/residual branches from the hot loop.
// M multiple of 128, N multiple of BN, K multiple of 32 by construction.
// ---------------------------------------------------------------------------
template<int BN, bool GATHER, bool SCATTER, bool EXPERT, bool RESID>
__global__ __launch_bounds__(256) void wmma_gemm(
    const float* __restrict__ A, int lda, long long sAz,
    const float* __restrict__ Bm, int ldb, long long sBz,
    float* __restrict__ Cm, int ldc, long long sCz,
    int K, float alpha,
    const float* __restrict__ Res, int ldres,
    const int* __restrict__ gidx,      // gather A rows (GATHER)
    const int* __restrict__ sidx,      // scatter C rows (SCATTER)
    const int* __restrict__ tileExp,   // per-M-tile expert id (EXPERT)
    long long sBe)                     // expert stride for B
{
  constexpr int FN = BN / 32;          // 16-wide fragments per wave in N
  const int mt = blockIdx.y, nt = blockIdx.x, z = blockIdx.z;
  const float* Ab = A + (long long)z * sAz;
  const float* Bb;
  if constexpr (EXPERT) {
    int e = tileExp[mt];
    if (e < 0) return;
    Bb = Bm + (long long)e * sBe;
  } else {
    Bb = Bm + (long long)z * sBz;
  }
  float* Cb = Cm + (long long)z * sCz;
  const int m0 = mt * 128, n0 = nt * BN;

  __shared__ _Float16 As [128][40];    // A tile, row-major (m,k)
  __shared__ _Float16 Bts[BN][40];     // B tile, transposed (n,k)

  const int tid  = threadIdx.x;
  const int w    = tid >> 5;
  const int lane = tid & 31;
  const int lm   = lane & 15;
  const int hi8  = (lane >= 16) ? 8  : 0;   // A k-offset / C m-offset
  const int kb16 = (lane >= 16) ? 16 : 0;   // unused; B k-offset is hi8 (kb = ka)
  const int wm   = (w >> 1) * 32;           // wave row in block tile
  const int wn   = (w & 1) * (BN / 2);      // wave col in block tile

  F32x8 cf[2][FN];
#pragma unroll
  for (int i = 0; i < 2; ++i)
#pragma unroll
    for (int j = 0; j < FN; ++j)
#pragma unroll
      for (int v = 0; v < 8; ++v) cf[i][j].f[v] = 0.0f;

  float4 aReg[4];        // 128x32 tile = 1024 float4 chunks / 256 threads
  float4 bReg[FN];       // 32xBN tile = 8*BN chunks / 256 threads = FN

  auto loadA = [&](int k0) {
#pragma unroll
    for (int j = 0; j < 4; ++j) {
      int idx4 = tid + j * 256;
      int row  = idx4 >> 3;
      int col  = (idx4 & 7) * 4;
      long long srow;
      if constexpr (GATHER) {
        int g = gidx[m0 + row];
        if (g < 0) { aReg[j] = make_float4(0.f, 0.f, 0.f, 0.f); continue; }
        srow = g;
      } else {
        srow = m0 + row;
      }
      aReg[j] = *(const float4*)(Ab + srow * lda + k0 + col);
    }
  };
  auto loadB = [&](int k0) {
#pragma unroll
    for (int j = 0; j < FN; ++j) {
      int idx4 = tid + j * 256;
      int kr   = idx4 / (BN / 4);
      int col  = (idx4 % (BN / 4)) * 4;
      bReg[j] = *(const float4*)(Bb + (long long)(k0 + kr) * ldb + n0 + col);
    }
  };
  auto storeTiles = [&]() {
#pragma unroll
    for (int j = 0; j < 4; ++j) {
      int idx4 = tid + j * 256;
      int row  = idx4 >> 3;
      int col  = (idx4 & 7) * 4;
      As[row][col + 0] = (_Float16)aReg[j].x;
      As[row][col + 1] = (_Float16)aReg[j].y;
      As[row][col + 2] = (_Float16)aReg[j].z;
      As[row][col + 3] = (_Float16)aReg[j].w;
    }
#pragma unroll
    for (int j = 0; j < FN; ++j) {
      int idx4 = tid + j * 256;
      int kr   = idx4 / (BN / 4);
      int col  = (idx4 % (BN / 4)) * 4;
      Bts[col + 0][kr] = (_Float16)bReg[j].x;
      Bts[col + 1][kr] = (_Float16)bReg[j].y;
      Bts[col + 2][kr] = (_Float16)bReg[j].z;
      Bts[col + 3][kr] = (_Float16)bReg[j].w;
    }
  };

  loadA(0);
  loadB(0);
  for (int k0 = 0; k0 < K; k0 += 32) {
    storeTiles();
    __syncthreads();
    if (k0 + 32 < K) {        // register double-buffer: prefetch next tiles
      loadA(k0 + 32);
      loadB(k0 + 32);
      if (k0 + 64 < K)        // gfx1250 global_prefetch of tile after next
        __builtin_prefetch((const void*)(Bb + (long long)(k0 + 64) * ldb + n0), 0, 1);
    }
    // ---- fragments + 2*FN WMMAs from LDS ----
    F16x16 af[2], bf[FN];
#pragma unroll
    for (int v = 0; v < 8; ++v) {
      int ka = ((v < 4) ? 2 * v : 16 + 2 * (v - 4)) + hi8;
      af[0].u[v] = *(const unsigned*)&As[wm + lm][ka];
      af[1].u[v] = *(const unsigned*)&As[wm + 16 + lm][ka];
      int kb = ka;
#pragma unroll
      for (int j = 0; j < FN; ++j)
        bf[j].u[v] = *(const unsigned*)&Bts[wn + j * 16 + lm][kb];
    }
#pragma unroll
    for (int j = 0; j < FN; ++j) {
      cf[0][j].v = wmma_f16(af[0].h, bf[j].h, cf[0][j].v);
      cf[1][j].v = wmma_f16(af[1].h, bf[j].h, cf[1][j].v);
    }
    __syncthreads();
  }

  // ---- epilogue: alpha, residual add, optional row scatter ----
#pragma unroll
  for (int i = 0; i < 2; ++i)
#pragma unroll
    for (int v = 0; v < 8; ++v) {
      int rrow = m0 + wm + i * 16 + v + hi8;
      long long orow;
      if constexpr (SCATTER) {
        int s = sidx[rrow];
        if (s < 0) continue;
        orow = s;
      } else {
        orow = rrow;
      }
#pragma unroll
      for (int j = 0; j < FN; ++j) {
        int col = n0 + wn + j * 16 + lm;
        float val = alpha * cf[i][j].f[v];
        if constexpr (RESID) val += Res[orow * ldres + col];
        Cb[orow * ldc + col] = val;
      }
    }
}

// ---------------------------------------------------------------------------
// RMSNorm: one block per token row of length C=1024
// ---------------------------------------------------------------------------
__global__ __launch_bounds__(256) void rmsnorm_kernel(
    const float* __restrict__ X, const float* __restrict__ w,
    float* __restrict__ Y)
{
  const int row = blockIdx.x;
  const float* x = X + (long long)row * CC;
  float acc = 0.f;
  for (int c = threadIdx.x; c < CC; c += 256) { float v = x[c]; acc += v * v; }
  __shared__ float red[256];
  red[threadIdx.x] = acc;
  __syncthreads();
  for (int st = 128; st > 0; st >>= 1) {
    if (threadIdx.x < st) red[threadIdx.x] += red[threadIdx.x + st];
    __syncthreads();
  }
  float rs = rsqrtf(red[0] * (1.0f / CC) + 1e-6f);
  for (int c = threadIdx.x; c < CC; c += 256)
    Y[(long long)row * CC + c] = x[c] * rs * w[c];
}

// ---------------------------------------------------------------------------
// RoPE (in place) on (B,T,H,D) buffer; one thread per rotation pair
// ---------------------------------------------------------------------------
__global__ __launch_bounds__(256) void rope_kernel(float* __restrict__ buf)
{
  long long i = (long long)blockIdx.x * 256 + threadIdx.x;  // NTOK*H*(D/2)
  if (i >= (long long)NTOK * HH * (DD / 2)) return;
  int j = (int)(i & 31);
  int h = (int)((i >> 5) & 15);
  long long row = i >> 9;
  int t = (int)(row & (TT - 1));
  float inv = __powf(100000.0f, -(float)(2 * j) / (float)DD);
  float f = (float)t * inv;
  float s, c;
  __sincosf(f, &s, &c);
  float* p = buf + row * CC + h * DD;
  float x1 = p[j], x2 = p[j + 32];
  p[j]      = x1 * c - x2 * s;
  p[j + 32] = x1 * s + x2 * c;
}

// ---------------------------------------------------------------------------
// Flash-style causal attention. Block = 8 waves, each wave owns 16 queries.
// K/V staged in LDS (f16); S^T = K*Q^T via WMMA (keys on M so per-query
// softmax stats reduce with a single shfl_xor(16)); P restaged through
// per-wave LDS into A-fragment layout for the P*V WMMA.
// grid = (T/128, H, B)
// ---------------------------------------------------------------------------
__global__ __launch_bounds__(256) void attn_kernel(
    const float* __restrict__ Q, const float* __restrict__ Kg,
    const float* __restrict__ Vg, float* __restrict__ O)
{
  const int b = blockIdx.z, h = blockIdx.y;
  const int qb0 = blockIdx.x * 128;

  __shared__ _Float16 Ks[32][72];       // keys x dims
  __shared__ _Float16 Vt[64][40];       // dims x keys (transposed)
  __shared__ _Float16 Ps[8][16][40];    // per-wave P: queries x keys
  __shared__ float mS[8][16], lS[8][16], cS[8][16];

  const int tid  = threadIdx.x;
  const int w    = tid >> 5;
  const int lane = tid & 31;
  const int lm   = lane & 15;
  const int hi8  = (lane >= 16) ? 8  : 0;
  const int kb16 = (lane >= 16) ? 16 : 0;
  const int qbase = qb0 + w * 16;

  // Q fragments (B operand of S^T), pre-scaled by 1/sqrt(D)
  F16x16 bq[2];
  {
    const float* qp = Q + ((long long)(b * TT + qbase + lm)) * CC + h * DD;
#pragma unroll
    for (int v = 0; v < 8; ++v) {
      int k0 = ((v < 4) ? 2 * v : 16 + 2 * (v - 4)) + hi8;
      bq[0].h[2 * v]     = (_Float16)(qp[k0]          * 0.125f);
      bq[0].h[2 * v + 1] = (_Float16)(qp[k0 + 1]      * 0.125f);
      bq[1].h[2 * v]     = (_Float16)(qp[32 + k0]     * 0.125f);
      bq[1].h[2 * v + 1] = (_Float16)(qp[32 + k0 + 1] * 0.125f);
    }
  }
  if (lane < 16) { mS[w][lm] = -3.0e38f; lS[w][lm] = 0.0f; }

  F32x8 of[4];
#pragma unroll
  for (int j = 0; j < 4; ++j)
#pragma unroll
    for (int v = 0; v < 8; ++v) of[j].f[v] = 0.0f;

  const int send = qb0 + 128;
  for (int s0 = 0; s0 < send; s0 += 32) {
    // ---- cooperative K/V tile staging (issue both loads, then convert) ----
    float4 kv4[2], vv4[2];
#pragma unroll
    for (int j = 0; j < 2; ++j) {
      int idx4 = tid + j * 256;
      int key  = idx4 >> 4;
      int col  = (idx4 & 15) * 4;
      long long rb = ((long long)(b * TT + s0 + key)) * CC + h * DD + col;
      kv4[j] = *(const float4*)(Kg + rb);
      vv4[j] = *(const float4*)(Vg + rb);
    }
#pragma unroll
    for (int j = 0; j < 2; ++j) {
      int idx4 = tid + j * 256;
      int key  = idx4 >> 4;
      int col  = (idx4 & 15) * 4;
      Ks[key][col + 0] = (_Float16)kv4[j].x;
      Ks[key][col + 1] = (_Float16)kv4[j].y;
      Ks[key][col + 2] = (_Float16)kv4[j].z;
      Ks[key][col + 3] = (_Float16)kv4[j].w;
      Vt[col + 0][key] = (_Float16)vv4[j].x;
      Vt[col + 1][key] = (_Float16)vv4[j].y;
      Vt[col + 2][key] = (_Float16)vv4[j].z;
      Vt[col + 3][key] = (_Float16)vv4[j].w;
    }
    __syncthreads();

    if (s0 <= qbase + 15) {    // wave-uniform: skip fully-masked key blocks
      // ---- S^T = K * Q^T : two 16x16 frags (keys on M, queries on N) ----
      F32x8 sf[2];
#pragma unroll
      for (int fi = 0; fi < 2; ++fi) {
        F16x16 a0, a1;
#pragma unroll
        for (int v = 0; v < 8; ++v) {
          int ka = ((v < 4) ? 2 * v : 16 + 2 * (v - 4)) + hi8;
          a0.u[v] = *(const unsigned*)&Ks[fi * 16 + lm][ka];
          a1.u[v] = *(const unsigned*)&Ks[fi * 16 + lm][32 + ka];
        }
        F32x8 zz;
#pragma unroll
        for (int v = 0; v < 8; ++v) zz.f[v] = 0.0f;
        zz.v = wmma_f16(a0.h, bq[0].h, zz.v);
        sf[fi].v = wmma_f16(a1.h, bq[1].h, zz.v);
      }
      // ---- causal mask + running max ----
      const int qg = qbase + lm;
      float tmax = -3.0e38f;
#pragma unroll
      for (int fi = 0; fi < 2; ++fi)
#pragma unroll
        for (int v = 0; v < 8; ++v) {
          int key = s0 + fi * 16 + v + hi8;
          float sv = sf[fi].f[v];
          if (key > qg) sv = -3.0e38f;
          sf[fi].f[v] = sv;
          tmax = fmaxf(tmax, sv);
        }
      tmax = fmaxf(tmax, __shfl_xor(tmax, 16, 32));
      float mold = mS[w][lm];
      float mnew = fmaxf(mold, tmax);
      float corr = __expf(mold - mnew);
      float psum = 0.f;
#pragma unroll
      for (int fi = 0; fi < 2; ++fi)
#pragma unroll
        for (int v = 0; v < 8; ++v) {
          float pv = __expf(sf[fi].f[v] - mnew);
          sf[fi].f[v] = pv;
          psum += pv;
        }
      psum += __shfl_xor(psum, 16, 32);
      float lold = lS[w][lm];
      float lnew = lold * corr + psum;
      if (lane < 16) { mS[w][lm] = mnew; lS[w][lm] = lnew; cS[w][lm] = corr; }
      // ---- stage P (transposed) to per-wave LDS, f16 ----
#pragma unroll
      for (int fi = 0; fi < 2; ++fi)
#pragma unroll
        for (int v = 0; v < 8; ++v)
          Ps[w][lm][fi * 16 + v + hi8] = (_Float16)sf[fi].f[v];
      // ---- O = O*corr + P*V ----
      float cr[8];
#pragma unroll
      for (int v = 0; v < 8; ++v) cr[v] = cS[w][v + hi8];
      F16x16 pf;
#pragma unroll
      for (int v = 0; v < 8; ++v) {
        int ka = ((v < 4) ? 2 * v : 16 + 2 * (v - 4)) + hi8;
        pf.u[v] = *(const unsigned*)&Ps[w][lm][ka];
      }
#pragma unroll
      for (int j = 0; j < 4; ++j) {
        F16x16 vf;
#pragma unroll
        for (int v = 0; v < 8; ++v)
          vf.u[v] = *(const unsigned*)&Vt[j * 16 + lm][((v < 4) ? 2 * v : 16 + 2 * (v - 4)) + hi8];
        F32x8 t;
#pragma unroll
        for (int v = 0; v < 8; ++v) t.f[v] = of[j].f[v] * cr[v];
        of[j].v = wmma_f16(pf.h, vf.h, t.v);
      }
    }
    __syncthreads();
  }

  // ---- normalize and store O (B,T,H,D layout) ----
  float linv[8];
#pragma unroll
  for (int v = 0; v < 8; ++v) linv[v] = 1.0f / lS[w][v + hi8];
#pragma unroll
  for (int j = 0; j < 4; ++j)
#pragma unroll
    for (int v = 0; v < 8; ++v) {
      long long row = (long long)(b * TT + qbase + v + hi8);
      O[row * CC + h * DD + j * 16 + lm] = of[j].f[v] * linv[v];
    }
}

// ---------------------------------------------------------------------------
// Router: per-token logits over 8 experts, sigmoid, top-1 (first-max ties)
// ---------------------------------------------------------------------------
__global__ __launch_bounds__(256) void router_kernel(
    const float* __restrict__ X, const float* __restrict__ Wr,
    const float* __restrict__ rb, int* __restrict__ sel,
    float* __restrict__ P)
{
  const int t = blockIdx.x;
  float acc[EE];
#pragma unroll
  for (int e = 0; e < EE; ++e) acc[e] = 0.f;
  for (int c = threadIdx.x; c < CC; c += 256) {
    float xv = X[(long long)t * CC + c];
#pragma unroll
    for (int e = 0; e < EE; ++e) acc[e] += xv * Wr[c * EE + e];
  }
  __shared__ float red[256][EE];
#pragma unroll
  for (int e = 0; e < EE; ++e) red[threadIdx.x][e] = acc[e];
  __syncthreads();
  for (int st = 128; st > 0; st >>= 1) {
    if (threadIdx.x < st)
#pragma unroll
      for (int e = 0; e < EE; ++e) red[threadIdx.x][e] += red[threadIdx.x + st][e];
    __syncthreads();
  }
  if (threadIdx.x == 0) {
    float best = -1.f;
    int bi = 0;
    for (int e = 0; e < EE; ++e) {
      float lg = red[0][e] + rb[e];
      lg = fminf(fmaxf(lg, -50.f), 50.f);
      float pr = 1.0f / (1.0f + __expf(-lg));
      if (pr > best) { best = pr; bi = e; }
    }
    sel[t] = bi;
    P[t] = fminf(fmaxf(best, 1e-8f), 1.0f - 1e-8f);
  }
}

// ---------------------------------------------------------------------------
// Router finalize: count tokens per expert, 128-padded segments, build
// gather index (pidx, -1 padding) and per-M-tile expert map. Integer LDS
// atomics only -> output deterministic (each token written exactly once).
// ---------------------------------------------------------------------------
__global__ __launch_bounds__(256) void finalize_kernel(
    const int* __restrict__ sel, int* __restrict__ pidx,
    int* __restrict__ tileExp)
{
  __shared__ int cnt[EE], cur[EE];
  const int tid = threadIdx.x;
  if (tid < EE) cnt[tid] = 0;
  __syncthreads();
  for (int t = tid; t < NTOK; t += 256) atomicAdd(&cnt[sel[t]], 1);
  __syncthreads();
  if (tid == 0) {
    int o = 0;
    for (int e = 0; e < EE; ++e) {
      cur[e] = o;
      int tiles = (cnt[e] + 127) >> 7;
      for (int tt2 = 0; tt2 < tiles; ++tt2) tileExp[(o >> 7) + tt2] = e;
      o += tiles << 7;
    }
    for (int tt2 = o >> 7; tt2 < NTILE; ++tt2) tileExp[tt2] = -1;
  }
  __syncthreads();
  for (int i = tid; i < NPAD; i += 256) pidx[i] = -1;
  __syncthreads();
  for (int t = tid; t < NTOK; t += 256) {
    int pos = atomicAdd(&cur[sel[t]], 1);
    pidx[pos] = t;
  }
}

// ---------------------------------------------------------------------------
// Elementwise: a = silu(a) * b   (in place into a)
// ---------------------------------------------------------------------------
__global__ __launch_bounds__(256) void silu_mul_kernel(
    float* __restrict__ a, const float* __restrict__ b, long long n)
{
  long long i = (long long)blockIdx.x * 256 + threadIdx.x;
  if (i < n) {
    float x = a[i];
    a[i] = x / (1.0f + __expf(-x)) * b[i];
  }
}

// ---------------------------------------------------------------------------
// Final MoE combine: out = out1 + (0.5*shared + p*y) / tw
// ---------------------------------------------------------------------------
__global__ __launch_bounds__(256) void combine_kernel(
    const float* __restrict__ out1, const float* __restrict__ sh,
    const float* __restrict__ y, const float* __restrict__ P,
    float* __restrict__ out)
{
  long long i = (long long)blockIdx.x * 256 + threadIdx.x;
  if (i >= (long long)NTOK * CC) return;
  int t = (int)(i >> 10);
  float p = P[t];
  float tw = fminf(fmaxf(0.5f + p + 1e-8f, 0.5f), 2.0f);
  out[i] = out1[i] + (0.5f * sh[i] + p * y[i]) / tw;
}

// ---------------------------------------------------------------------------
// Host orchestration
// ---------------------------------------------------------------------------
extern "C" void kernel_launch(void* const* d_in, const int* in_sizes, int n_in,
                              void* d_out, int out_size, void* d_ws, size_t ws_size,
                              hipStream_t stream) {
  (void)in_sizes; (void)n_in; (void)out_size; (void)ws_size;
  const float* x     = (const float*)d_in[0];
  const float* ln1   = (const float*)d_in[1];
  const float* ln2   = (const float*)d_in[2];
  const float* Wq    = (const float*)d_in[3];
  const float* Wkv   = (const float*)d_in[4];
  const float* Wk_up = (const float*)d_in[5];
  const float* Wv_up = (const float*)d_in[6];
  const float* Wo    = (const float*)d_in[7];
  const float* Wr    = (const float*)d_in[8];
  const float* rb    = (const float*)d_in[9];
  const float* sw1   = (const float*)d_in[10];
  const float* sw2   = (const float*)d_in[11];
  const float* sw3   = (const float*)d_in[12];
  const float* ew1   = (const float*)d_in[13];
  const float* ew2   = (const float*)d_in[14];
  const float* ew3   = (const float*)d_in[15];
  float* out = (float*)d_out;
  float* ws  = (float*)d_ws;

  const long long M1 = (long long)NTOK * CC;     // 8,388,608 floats
  // phase-aliased arena (floats)
  float* out1 = ws;                  // kept until the end
  float* r    = ws + M1;             // reuse region
  // attention phase
  float* xn   = r;
  float* qb   = r + M1;
  float* kv   = r + 2 * M1;
  float* kb   = kv + (long long)NTOK * LL;
  float* vb   = kb + M1;
  float* at   = vb + M1;
  // moe phase (aliases attention temps)
  float* xn2  = r;
  float* h1   = r + M1;                          // NTOK*I, holds hs in place
  float* h3   = h1 + (long long)NTOK * II;
  float* eh1  = h3;                              // reuses h3 space after silu
  float* eh3  = eh1 + (long long)NPAD * II;
  float* sh   = eh3 + (long long)NPAD * II;
  float* yb   = sh + M1;
  float* smallf = yb + M1;
  float* Pp     = smallf;                        // 8192 floats
  int*   sel    = (int*)(smallf + NTOK);
  int*   pidx   = sel + NTOK;
  int*   tileExp= pidx + NPAD;

  const dim3 blk(256);

  // ---- attention ----
  rmsnorm_kernel<<<NTOK, blk, 0, stream>>>(x, ln1, xn);
  // q = xn @ Wq
  wmma_gemm<128, false, false, false, false><<<dim3(CC / 128, NTOK / 128, 1), blk, 0, stream>>>(
      xn, CC, 0, Wq, CC, 0, qb, CC, 0, CC, 1.0f,
      nullptr, 0, nullptr, nullptr, nullptr, 0);
  // kv = xn @ Wkv * 1/sqrt(H)
  wmma_gemm<128, false, false, false, false><<<dim3(LL / 128, NTOK / 128, 1), blk, 0, stream>>>(
      xn, CC, 0, Wkv, LL, 0, kv, LL, 0, CC, 0.25f,
      nullptr, 0, nullptr, nullptr, nullptr, 0);
  // k = kv @ Wk_up[h]  (batched over heads; output columns h*D..)
  wmma_gemm<64, false, false, false, false><<<dim3(1, NTOK / 128, HH), blk, 0, stream>>>(
      kv, LL, 0, Wk_up, DD, (long long)LL * DD, kb, CC, DD, LL, 1.0f,
      nullptr, 0, nullptr, nullptr, nullptr, 0);
  // v = kv @ Wv_up[h]
  wmma_gemm<64, false, false, false, false><<<dim3(1, NTOK / 128, HH), blk, 0, stream>>>(
      kv, LL, 0, Wv_up, DD, (long long)LL * DD, vb, CC, DD, LL, 1.0f,
      nullptr, 0, nullptr, nullptr, nullptr, 0);
  // rope on q, k
  const long long nrope = (long long)NTOK * HH * (DD / 2);
  rope_kernel<<<(unsigned)((nrope + 255) / 256), blk, 0, stream>>>(qb);
  rope_kernel<<<(unsigned)((nrope + 255) / 256), blk, 0, stream>>>(kb);
  // flash attention
  attn_kernel<<<dim3(TT / 128, HH, BB), blk, 0, stream>>>(qb, kb, vb, at);
  // out1 = x + attn_out @ Wo
  wmma_gemm<128, false, false, false, true><<<dim3(CC / 128, NTOK / 128, 1), blk, 0, stream>>>(
      at, CC, 0, Wo, CC, 0, out1, CC, 0, CC, 1.0f,
      x, CC, nullptr, nullptr, nullptr, 0);

  // ---- MoE ----
  rmsnorm_kernel<<<NTOK, blk, 0, stream>>>(out1, ln2, xn2);
  router_kernel<<<NTOK, blk, 0, stream>>>(xn2, Wr, rb, sel, Pp);
  finalize_kernel<<<1, blk, 0, stream>>>(sel, pidx, tileExp);
  // shared expert swiglu
  wmma_gemm<128, false, false, false, false><<<dim3(II / 128, NTOK / 128, 1), blk, 0, stream>>>(
      xn2, CC, 0, sw1, II, 0, h1, II, 0, CC, 1.0f,
      nullptr, 0, nullptr, nullptr, nullptr, 0);
  wmma_gemm<128, false, false, false, false><<<dim3(II / 128, NTOK / 128, 1), blk, 0, stream>>>(
      xn2, CC, 0, sw3, II, 0, h3, II, 0, CC, 1.0f,
      nullptr, 0, nullptr, nullptr, nullptr, 0);
  silu_mul_kernel<<<(unsigned)(((long long)NTOK * II + 255) / 256), blk, 0, stream>>>(
      h1, h3, (long long)NTOK * II);
  wmma_gemm<128, false, false, false, false><<<dim3(CC / 128, NTOK / 128, 1), blk, 0, stream>>>(
      h1, II, 0, sw2, CC, 0, sh, CC, 0, II, 1.0f,
      nullptr, 0, nullptr, nullptr, nullptr, 0);
  // switched experts: gathered rows, per-tile expert weights
  wmma_gemm<128, true, false, true, false><<<dim3(II / 128, NTILE, 1), blk, 0, stream>>>(
      xn2, CC, 0, ew1, II, 0, eh1, II, 0, CC, 1.0f,
      nullptr, 0, pidx, nullptr, tileExp, (long long)CC * II);
  wmma_gemm<128, true, false, true, false><<<dim3(II / 128, NTILE, 1), blk, 0, stream>>>(
      xn2, CC, 0, ew3, II, 0, eh3, II, 0, CC, 1.0f,
      nullptr, 0, pidx, nullptr, tileExp, (long long)CC * II);
  silu_mul_kernel<<<(unsigned)(((long long)NPAD * II + 255) / 256), blk, 0, stream>>>(
      eh1, eh3, (long long)NPAD * II);
  // down-projection with scatter back to token rows
  wmma_gemm<128, false, true, true, false><<<dim3(CC / 128, NTILE, 1), blk, 0, stream>>>(
      eh1, II, 0, ew2, CC, 0, yb, CC, 0, II, 1.0f,
      nullptr, 0, nullptr, pidx, tileExp, (long long)II * CC);
  // final combine
  combine_kernel<<<(unsigned)((M1 + 255) / 256), blk, 0, stream>>>(
      out1, sh, yb, Pp, out);
}
